// QRNNLayer_noncausal_43009802502835
// MI455X (gfx1250) — matrix operand order repaired and checked by reading, not compiled, not run on hardware
//
#include <hip/hip_runtime.h>
#include <hip/hip_bf16.h>

#define BB   16
#define TT   8192
#define CC   128
#define HH   128
#define KT   5          // conv taps
#define T1   8193       // T+1 (prepended init state row)
#define NC   64         // scan chunks along time
#define CL   129        // chunk length (64*129 >= 8193)
#define NEG_INF (-__builtin_inff())

typedef __attribute__((ext_vector_type(16))) __bf16 v16bf;
typedef __attribute__((ext_vector_type(8)))  float  v8f;
typedef __attribute__((ext_vector_type(4)))  unsigned int u32x4;
typedef __attribute__((ext_vector_type(8)))  int          i32x8;
typedef __attribute__((ext_vector_type(4)))  int          i32x4;

union Frag32B { uint4 u4[2]; v16bf v; };

static __device__ __forceinline__ unsigned short f2bf(float f) {
    unsigned int u = __float_as_uint(f);
    unsigned int r = u + 0x7FFFu + ((u >> 16) & 1u);   // round-to-nearest-even
    return (unsigned short)(r >> 16);
}

static __device__ __forceinline__ float logsig(float x) {
    // log(sigmoid(x)) = min(x,0) - log1p(exp(-|x|))
    return fminf(x, 0.0f) - log1pf(__expf(-fabsf(x)));
}

static __device__ __forceinline__ float logaddexp(float a, float b) {
    float m = fmaxf(a, b);
    if (m == NEG_INF) return NEG_INF;
    return m + __logf(__expf(a - m) + __expf(b - m));
}

// ---------------------------------------------------------------------------
// One-shot fp32 -> bf16 conversion of the inputs so the conv kernel can DMA
// tiles straight into LDS with the Tensor Data Mover (no per-tile VALU work).
// ---------------------------------------------------------------------------
__global__ __launch_bounds__(256)
void x2bf_kernel(const float* __restrict__ x, unsigned short* __restrict__ xbf) {
    size_t i = ((size_t)blockIdx.x * 256 + threadIdx.x) * 4;   // B*T*C / 4 threads
    float4 v = *(const float4*)(x + i);
    ushort4 o;
    o.x = f2bf(v.x); o.y = f2bf(v.y); o.z = f2bf(v.z); o.w = f2bf(v.w);
    *(ushort4*)(xbf + i) = o;
}

// ---------------------------------------------------------------------------
// Pack weights (H,C,K) fp32 -> bf16 laid out exactly as WMMA B-fragments.
// Fragment layout (16-bit B 32x16, contiguous k order matching the A loads): lane -> column N, element e ->
// K = e + 16*(lane>=16).  Linear index:
//   (((half*8+nt)*20 + kstep)*32 + lane)*16 + e
// kstep = tap*4 + cchunk ; c = cchunk*32 + K ; h = nt*16 + lane%16.
// ---------------------------------------------------------------------------
__global__ __launch_bounds__(256)
void pack_w_kernel(const float* __restrict__ zw, const float* __restrict__ fw,
                   unsigned short* __restrict__ wpk) {
    int idx = blockIdx.x * 256 + threadIdx.x;      // 163840 total
    if (idx >= 2 * 8 * 20 * 32 * 16) return;
    int e     = idx & 15;
    int lane  = (idx >> 4) & 31;
    int kstep = (idx >> 9) % 20;
    int nth   = idx / (16 * 32 * 20);              // 0..15
    int nt    = nth & 7;
    int half  = nth >> 3;
    int h     = nt * 16 + (lane & 15);
    int Kr    = e + ((lane >> 4) << 4);
    int tap   = kstep >> 2;
    int c     = (kstep & 3) * 32 + Kr;
    const float* w = half ? fw : zw;
    wpk[idx] = f2bf(w[(h * CC + c) * KT + tap]);
}

// Row 0 of raw buffers = prepended init state (no bias, per reference).
__global__ __launch_bounds__(256)
void init_row_kernel(const float* __restrict__ init,
                     float* __restrict__ rawZ, float* __restrict__ rawF) {
    int idx = blockIdx.x * 256 + threadIdx.x;      // B*H = 2048
    if (idx >= BB * HH) return;
    int b = idx / HH, h = idx % HH;
    rawF[((size_t)b * T1) * HH + h] = init[b * HH + h];            // init_state[0]
    rawZ[((size_t)b * T1) * HH + h] = init[BB * HH + b * HH + h];  // init_state[1]
}

// ---------------------------------------------------------------------------
// Fused conv1d(z) + conv1d(f) as 5 shifted bf16 GEMMs via v_wmma_f32_16x16x32_bf16.
// A-tile (64 rows + 4 halo, 128 channels, bf16) staged into LDS by the Tensor
// Data Mover with hardware row padding (256B payload + 16B pad = 272B stride).
// Block: 64 time rows x 256 cols (z|f). 8 waves: (mtile 0..3) x (half 0..1).
// ---------------------------------------------------------------------------
__global__ __launch_bounds__(256)
void conv_wmma_kernel(const unsigned short* __restrict__ xbf,
                      const unsigned short* __restrict__ wpk,
                      const float* __restrict__ zb, const float* __restrict__ fb,
                      float* __restrict__ rawZ, float* __restrict__ rawF) {
    __shared__ unsigned short sx[68 * 136];        // 68 rows, 272B stride (16B pad)

    const int b   = blockIdx.y;
    const int t0  = blockIdx.x * 64;
    const int tid = threadIdx.x;

    // TDM tile: clamped halo row range [g0row, endRow) of this batch's (T,C) slab.
    const int g0row   = (t0 == 0) ? 0 : (t0 - 2);
    const int endRow  = (t0 + 66 > TT) ? TT : (t0 + 66);
    const int nrows   = endRow - g0row;            // 66 or 68
    const int ldsRow0 = (t0 == 0) ? 2 : 0;

    if (tid < 32) {                                // single wave issues the DMA
        unsigned ldsAddr = (unsigned)(size_t)(void*)sx + (unsigned)(ldsRow0 * 272);
        unsigned long long ga =
            (unsigned long long)(size_t)(xbf + ((size_t)b * TT + g0row) * CC);
        u32x4 g0;
        g0[0] = 1u;                                        // count=1 (valid D#)
        g0[1] = ldsAddr;                                   // lds_addr
        g0[2] = (unsigned)(ga & 0xFFFFFFFFull);            // global_addr[31:0]
        g0[3] = (unsigned)((ga >> 32) & 0x01FFFFFFull)     // global_addr[56:32]
              | (2u << 30);                                // type=2 ("image")
        i32x8 g1;
        g1[0] = (int)((1u << 16)     // data_size = 2 bytes
                    | (1u << 20)     // pad_enable
                    | (5u << 22)     // pad_interval: every 64 DWORDs (256B)
                    | (3u << 25));   // pad_amount: 4 DWORDs (16B)
        g1[1] = (int)(128u << 16);                 // tensor_dim0 = 128 (low16)
        g1[2] = (int)(((unsigned)TT & 0xFFFFu) << 16); // tensor_dim1 = T (low16)
        g1[3] = (int)(128u << 16);                 // tile_dim0 = 128
        g1[4] = nrows;                             // tile_dim1
        g1[5] = 128;                               // tensor_dim0_stride
        g1[6] = 0;
        g1[7] = 0;
        i32x4 gz = {0, 0, 0, 0};                   // 2-D tile: groups 2/3 unused
#if __has_include(<hip/amd_detail/amd_gfx1250_TDM.h>)
        i32x8 gz8 = {0, 0, 0, 0, 0, 0, 0, 0};      // amdgpu-toolchain: 6-arg form
        __builtin_amdgcn_tensor_load_to_lds(g0, g1, gz, gz, gz8, 0);
#else
        __builtin_amdgcn_tensor_load_to_lds(g0, g1, gz, gz, 0);  // ROCm 7.2: 5-arg
#endif
    }

    // Zero-fill halo rows the TDM tile does not cover (batch time edges only).
    if (t0 == 0)
        for (int i = tid; i < 2 * 136; i += 256) sx[i] = 0;
    if (t0 + 66 > TT)
        for (int i = tid; i < 2 * 136; i += 256) sx[66 * 136 + i] = 0;

    __builtin_amdgcn_s_wait_tensorcnt(0);          // no-op for waves 1..7
    __syncthreads();

    const int lane  = tid & 31;
    const int wid   = tid >> 5;
    const int mtile = wid & 3;
    const int half  = wid >> 2;                    // 0 = z, 1 = f
    const int lhi   = lane >> 4;
    const int lmod  = lane & 15;

    v8f acc[8];
    #pragma unroll
    for (int i = 0; i < 8; ++i) acc[i] = (v8f){0,0,0,0,0,0,0,0};

    const int rowA = mtile * 16 + lmod;            // local time row before tap shift

    for (int tap = 0; tap < KT; ++tap) {
        #pragma unroll
        for (int cc4 = 0; cc4 < 4; ++cc4) {
            const int kstep = tap * 4 + cc4;
            // A fragment (contiguous layout: lanes<16 K=0..15; lanes>=16 K=16..31)
            const unsigned short* ap = &sx[(rowA + tap) * 136 + cc4 * 32 + lhi * 16];
            Frag32B af;
            af.u4[0] = *(const uint4*)(ap);
            af.u4[1] = *(const uint4*)(ap + 8);
            #pragma unroll
            for (int nt = 0; nt < 8; ++nt) {
                const unsigned short* bp =
                    wpk + ((((size_t)(half * 8 + nt) * 20 + kstep) * 32 + lane) * 16);
                Frag32B bf_;
                bf_.u4[0] = *(const uint4*)(bp);
                bf_.u4[1] = *(const uint4*)(bp + 8);
                acc[nt] = __builtin_amdgcn_wmma_f32_16x16x32_bf16(
                    false, af.v, false, bf_.v, (short)0, acc[nt], false, false);
            }
        }
    }

    float*       raw  = half ? rawF : rawZ;
    const float* bias = half ? fb   : zb;
    #pragma unroll
    for (int nt = 0; nt < 8; ++nt) {
        const int h  = nt * 16 + lmod;
        const float bv = bias[h];
        #pragma unroll
        for (int r = 0; r < 8; ++r) {
            const int m = r + lhi * 8;             // C/D layout: VGPR r, lanes>=16 -> M+8
            const int t = t0 + mtile * 16 + m;
            raw[((size_t)b * T1 + (t + 1)) * HH + h] = acc[nt][r] + bv;
        }
    }
}

// -------------------------- chunked scan passes ----------------------------
__global__ __launch_bounds__(128)
void scanA_kernel(const float* __restrict__ rawF, float* __restrict__ chunkSum) {
    int b = blockIdx.x / NC, c = blockIdx.x % NC, h = threadIdx.x;
    int lo = c * CL, hi = lo + CL; if (hi > T1) hi = T1;
    float s = 0.0f;
    for (int t = lo; t < hi; ++t)
        s += logsig(rawF[((size_t)b * T1 + t) * HH + h]);
    chunkSum[((size_t)b * NC + c) * HH + h] = s;
}

__global__ __launch_bounds__(128)
void scanB_kernel(const float* __restrict__ chunkSum, float* __restrict__ sufAfter) {
    int b = blockIdx.x, h = threadIdx.x;
    float run = 0.0f;
    for (int c = NC - 1; c >= 0; --c) {
        size_t i = ((size_t)b * NC + c) * HH + h;
        sufAfter[i] = run;
        run += chunkSum[i];
    }
}

__global__ __launch_bounds__(128)
void scanC_kernel(float* __restrict__ rawZ, float* __restrict__ rawF,
                  const float* __restrict__ sufAfter, float* __restrict__ chunkLse) {
    int b = blockIdx.x / NC, c = blockIdx.x % NC, h = threadIdx.x;
    int lo = c * CL, hi = lo + CL; if (hi > T1) hi = T1;
    float run = sufAfter[((size_t)b * NC + c) * HH + h];
    float mx = NEG_INF, sm = 0.0f;                 // streaming logsumexp state
    for (int t = hi - 1; t >= lo; --t) {
        size_t i = ((size_t)b * T1 + t) * HH + h;
        float rf = rawF[i], rz = rawZ[i];
        float lf = logsig(rf);
        float rg = lf + run;                        // suffix sum of log_f from t
        run += lf;
        float s = logsig(rz) + logsig(-rf) + rg;
        rawZ[i] = s;                                // reuse buffers in place
        rawF[i] = rg;
        if (s > mx) { sm = sm * __expf(mx - s) + 1.0f; mx = s; }
        else        { sm += __expf(s - mx); }
    }
    chunkLse[((size_t)b * NC + c) * HH + h] = mx + __logf(sm);
}

__global__ __launch_bounds__(128)
void scanD_kernel(const float* __restrict__ chunkLse, float* __restrict__ carry) {
    int b = blockIdx.x, h = threadIdx.x;
    float L = NEG_INF;
    for (int c = 0; c < NC; ++c) {
        size_t i = ((size_t)b * NC + c) * HH + h;
        carry[i] = L;                               // exclusive prefix logaddexp
        L = logaddexp(L, chunkLse[i]);
    }
}

__global__ __launch_bounds__(128)
void scanE_kernel(const float* __restrict__ rawZ, const float* __restrict__ rawF,
                  const float* __restrict__ carry, float* __restrict__ out) {
    int b = blockIdx.x / NC, c = blockIdx.x % NC, h = threadIdx.x;
    int lo = c * CL, hi = lo + CL; if (hi > T1) hi = T1;
    float L = carry[((size_t)b * NC + c) * HH + h];
    for (int t = lo; t < hi; ++t) {
        size_t i = ((size_t)b * T1 + t) * HH + h;
        L = logaddexp(L, rawZ[i]);                  // rawZ holds s
        out[i] = __expf(L - rawF[i]);               // rawF holds rg
    }
}

extern "C" void kernel_launch(void* const* d_in, const int* in_sizes, int n_in,
                              void* d_out, int out_size, void* d_ws, size_t ws_size,
                              hipStream_t stream) {
    const float* x    = (const float*)d_in[0];   // (B,T,C)
    const float* init = (const float*)d_in[1];   // (2,B,H)
    const float* zw   = (const float*)d_in[2];   // (H,C,K)
    const float* zb   = (const float*)d_in[3];   // (H)
    const float* fw   = (const float*)d_in[4];   // (H,C,K)
    const float* fb   = (const float*)d_in[5];   // (H)
    float* out = (float*)d_out;                  // (B,T+1,H)

    const size_t nRaw = (size_t)BB * T1 * HH;    // 16,779,264
    const size_t nX   = (size_t)BB * TT * CC;    // 16,777,216
    float*          rawZ     = (float*)d_ws;
    float*          rawF     = rawZ + nRaw;
    unsigned short* xbf      = (unsigned short*)(rawF + nRaw);     // bf16 inputs
    unsigned short* wpk      = xbf + nX;                           // 163840 u16
    float*          chunkSum = (float*)(wpk + 163840);
    float*          sufAfter = chunkSum + (size_t)BB * NC * HH;
    float*          chunkLse = sufAfter + (size_t)BB * NC * HH;
    float*          carry    = chunkLse + (size_t)BB * NC * HH;

    x2bf_kernel<<<(int)(nX / 1024), 256, 0, stream>>>(x, xbf);
    pack_w_kernel<<<640, 256, 0, stream>>>(zw, fw, wpk);
    init_row_kernel<<<8, 256, 0, stream>>>(init, rawZ, rawF);

    dim3 cgrid(TT / 64, BB);
    conv_wmma_kernel<<<cgrid, 256, 0, stream>>>(xbf, wpk, zb, fb, rawZ, rawF);

    scanA_kernel<<<BB * NC, 128, 0, stream>>>(rawF, chunkSum);
    scanB_kernel<<<BB, 128, 0, stream>>>(chunkSum, sufAfter);
    scanC_kernel<<<BB * NC, 128, 0, stream>>>(rawZ, rawF, sufAfter, chunkLse);
    scanD_kernel<<<BB, 128, 0, stream>>>(chunkLse, carry);
    scanE_kernel<<<BB * NC, 128, 0, stream>>>(rawZ, rawF, carry, out);
}
